// Attention_2001454760270
// MI455X (gfx1250) — matrix operand order repaired and checked by reading, not compiled, not run on hardware
//
#include <hip/hip_runtime.h>

typedef __attribute__((ext_vector_type(8)))  _Float16 v8h;
typedef __attribute__((ext_vector_type(16))) _Float16 v16h;
typedef __attribute__((ext_vector_type(8)))  float    v8f;

#define B_ 8
#define S_ 2048
#define E_ 1024
#define F_ 3072  // 3*E (packed Q|K|V features)

// ---------------------------------------------------------------------------
// fp32 -> fp16 convert (grid-stride)
// ---------------------------------------------------------------------------
__global__ void cvt_f32_f16(const float* __restrict__ src, _Float16* __restrict__ dst, int n) {
    int i = blockIdx.x * blockDim.x + threadIdx.x;
    int stride = gridDim.x * blockDim.x;
    for (; i < n; i += stride) dst[i] = (_Float16)src[i];
}

__global__ void copy_f32(const float* __restrict__ src, float* __restrict__ dst, int n) {
    int i = blockIdx.x * blockDim.x + threadIdx.x;
    if (i < n) dst[i] = src[i];
}

// ---------------------------------------------------------------------------
// WMMA fragment loads per CDNA5 ISA layouts (16-bit operands, wave32).
// A 16x32: lane m=L&15; halves 0..7 = K[hi*8 .. hi*8+7], halves 8..15 = K[16+hi*8 ..]
// B 32x16: lane n=L&15; halves 0..7 = K[hi*8 .. hi*8+7], halves 8..15 = K[16+hi*8 ..] (same as A)
// ---------------------------------------------------------------------------
__device__ __forceinline__ v16h load_fragA(const _Float16* rowk, int hi) {
    v8h lo = *(const v8h*)(rowk + hi * 8);
    v8h up = *(const v8h*)(rowk + 16 + hi * 8);
    v16h r;
#pragma unroll
    for (int i = 0; i < 8; ++i) { r[i] = lo[i]; r[i + 8] = up[i]; }
    return r;
}

__device__ __forceinline__ v16h load_fragB(const _Float16* rowk, int hi) {
    return load_fragA(rowk, hi);
}

// ---------------------------------------------------------------------------
// Generic batched GEMM:  C[z] = alpha * A[z] * B[z]^T (+ bias)
//   A: M x K rows (lda), B: N x K rows (ldb)  -> C: M x N (ldc)
// Block: 256 thr = 8 waves (2m x 4n); wave tile 64x64 => 16 WMMA accums.
// Block tile 128 x 256. __launch_bounds__(256,1): allocate for 1 wave/EU so
// the 128-VGPR accumulator file + fragments fit in 256 VGPRs without spills.
// All dims are exact multiples (no bounds checks; EXEC all-ones for WMMA).
// ---------------------------------------------------------------------------
template <bool HAS_BIAS, bool OUT_F32>
__global__ __launch_bounds__(256, 1)
void wmma_gemm(const _Float16* __restrict__ A, size_t strideA, int lda,
               const _Float16* __restrict__ Bm, size_t strideB, int ldb,
               const float* __restrict__ bias,
               void* __restrict__ Cout, size_t strideC, int ldc,
               int K, float alpha) {
    const int lane = threadIdx.x & 31;
    const int wave = threadIdx.x >> 5;
    const int hi   = (lane >> 4) & 1;
    const int l15  = lane & 15;
    const int wm   = wave >> 2;  // 0..1
    const int wn   = wave & 3;   // 0..3
    const int m0 = blockIdx.y * 128 + wm * 64;
    const int n0 = blockIdx.x * 256 + wn * 64;

    const _Float16* Ab = A + strideA * blockIdx.z;
    const _Float16* Bb = Bm + strideB * blockIdx.z;

    const _Float16* aRow[4];
    const _Float16* bRow[4];
#pragma unroll
    for (int i = 0; i < 4; ++i) aRow[i] = Ab + (size_t)(m0 + i * 16 + l15) * lda;
#pragma unroll
    for (int j = 0; j < 4; ++j) bRow[j] = Bb + (size_t)(n0 + j * 16 + l15) * ldb;

    v8f acc[4][4] = {};

    for (int kk = 0; kk < K; kk += 32) {
        // prefetch two K-steps ahead (global_prefetch_b8; speculative, no guard)
#pragma unroll
        for (int i = 0; i < 4; ++i) __builtin_prefetch(aRow[i] + kk + 64, 0, 1);
#pragma unroll
        for (int j = 0; j < 4; ++j) __builtin_prefetch(bRow[j] + kk + 64, 0, 1);

        v16h a[4], b[4];
#pragma unroll
        for (int i = 0; i < 4; ++i) a[i] = load_fragA(aRow[i] + kk, hi);
#pragma unroll
        for (int j = 0; j < 4; ++j) b[j] = load_fragB(bRow[j] + kk, hi);
#pragma unroll
        for (int i = 0; i < 4; ++i)
#pragma unroll
            for (int j = 0; j < 4; ++j)
                acc[i][j] = __builtin_amdgcn_wmma_f32_16x16x32_f16(
                    false, a[i], false, b[j], (short)0, acc[i][j], false, false);
    }

    // Store: D layout m = r + 8*hi, n = L&15. Straight-line (templated) epilogue.
    float*    C32 = (float*)Cout;
    _Float16* C16 = (_Float16*)Cout;
    const size_t cBase = strideC * blockIdx.z;
#pragma unroll
    for (int j = 0; j < 4; ++j) {
        const int col = n0 + j * 16 + l15;
        float bv = 0.0f;
        if (HAS_BIAS) bv = bias[col];
#pragma unroll
        for (int i = 0; i < 4; ++i) {
#pragma unroll
            for (int r = 0; r < 8; ++r) {
                const int row = m0 + i * 16 + r + hi * 8;
                const float v = alpha * acc[i][j][r] + bv;
                const size_t idx = cBase + (size_t)row * ldc + col;
                if (OUT_F32) C32[idx] = v;
                else         C16[idx] = (_Float16)v;
            }
        }
    }
}

// ---------------------------------------------------------------------------
// Transpose V slice of qkv: vt[b][e][s] = qkv[(b*S+s)*F + 2E + e]
// 32x32 tiles through LDS (bank-conflict-free via +1 pad).
// ---------------------------------------------------------------------------
__global__ void transpose_v(const _Float16* __restrict__ qkv, _Float16* __restrict__ vt) {
    __shared__ _Float16 tile[32][33];
    const int b  = blockIdx.z;
    const int s0 = blockIdx.x * 32;
    const int e0 = blockIdx.y * 32;
    const int tx = threadIdx.x;  // 0..31
    const int ty = threadIdx.y;  // 0..7
#pragma unroll
    for (int i = ty; i < 32; i += 8)
        tile[i][tx] = qkv[((size_t)(b * S_ + s0 + i)) * F_ + 2 * E_ + e0 + tx];
    __syncthreads();
#pragma unroll
    for (int i = ty; i < 32; i += 8)
        vt[((size_t)(b * E_ + e0 + i)) * S_ + s0 + tx] = tile[tx][i];
}

// ---------------------------------------------------------------------------
// Row softmax with reference's masked_fill(1e-9) pre-softmax (NOT -inf:
// masked logits become exactly 1e-9 and participate in max/sum).
// One 256-thread block per row of 2048.
// ---------------------------------------------------------------------------
__global__ void softmax_mask(_Float16* __restrict__ scores,
                             const unsigned char* __restrict__ mask) {
    const int row = blockIdx.x;       // b*S + q
    const int b   = row >> 11;        // row / 2048
    _Float16* p = scores + (size_t)row * S_;
    const unsigned char* mrow = mask + (size_t)b * S_;

    __shared__ float red[256];
    float vals[8];
    float lmax = -1e30f;
#pragma unroll
    for (int i = 0; i < 8; ++i) {
        const int k = threadIdx.x + i * 256;
        const float v = mrow[k] ? 1e-9f : (float)p[k];
        vals[i] = v;
        lmax = fmaxf(lmax, v);
    }
    red[threadIdx.x] = lmax;
    __syncthreads();
    for (int off = 128; off > 0; off >>= 1) {
        if (threadIdx.x < off)
            red[threadIdx.x] = fmaxf(red[threadIdx.x], red[threadIdx.x + off]);
        __syncthreads();
    }
    const float rmax = red[0];
    __syncthreads();

    float lsum = 0.0f;
#pragma unroll
    for (int i = 0; i < 8; ++i) {
        const float e = __expf(vals[i] - rmax);
        vals[i] = e;
        lsum += e;
    }
    red[threadIdx.x] = lsum;
    __syncthreads();
    for (int off = 128; off > 0; off >>= 1) {
        if (threadIdx.x < off)
            red[threadIdx.x] += red[threadIdx.x + off];
        __syncthreads();
    }
    const float inv = 1.0f / red[0];
#pragma unroll
    for (int i = 0; i < 8; ++i)
        p[threadIdx.x + i * 256] = (_Float16)(vals[i] * inv);
}

// ---------------------------------------------------------------------------
// Launch
// ---------------------------------------------------------------------------
extern "C" void kernel_launch(void* const* d_in, const int* in_sizes, int n_in,
                              void* d_out, int out_size, void* d_ws, size_t ws_size,
                              hipStream_t stream) {
    (void)in_sizes; (void)n_in; (void)out_size; (void)ws_size;
    const float* x  = (const float*)d_in[0];
    const float* Wq = (const float*)d_in[1];
    const float* bq = (const float*)d_in[2];
    const float* Wk = (const float*)d_in[3];
    const float* bk = (const float*)d_in[4];
    const float* Wv = (const float*)d_in[5];
    const float* bv = (const float*)d_in[6];
    const unsigned char* mask = (const unsigned char*)d_in[7];
    float* out = (float*)d_out;

    // Workspace carve (256B aligned): ~240 MB total
    char* ws = (char*)d_ws;
    size_t off = 0;
    auto carve = [&](size_t bytes) -> char* {
        char* p = ws + off;
        off = (off + bytes + 255) & ~(size_t)255;
        return p;
    };
    _Float16* x16   = (_Float16*)carve((size_t)B_ * S_ * E_ * 2);  // 32 MB
    _Float16* w16   = (_Float16*)carve((size_t)F_ * E_ * 2);       // 6 MB
    float*    biasP = (float*)   carve((size_t)F_ * 4);
    _Float16* qkv16 = (_Float16*)carve((size_t)B_ * S_ * F_ * 2);  // 96 MB
    _Float16* vt16  = (_Float16*)carve((size_t)B_ * E_ * S_ * 2);  // 32 MB
    _Float16* sc16  = (_Float16*)carve((size_t)B_ * S_ * S_ * 2);  // 64 MB

    // 1) precision conversion + weight/bias packing
    cvt_f32_f16<<<dim3(2048), dim3(256), 0, stream>>>(x,  x16, B_ * S_ * E_);
    cvt_f32_f16<<<dim3(512),  dim3(256), 0, stream>>>(Wq, w16,               E_ * E_);
    cvt_f32_f16<<<dim3(512),  dim3(256), 0, stream>>>(Wk, w16 +     E_ * E_, E_ * E_);
    cvt_f32_f16<<<dim3(512),  dim3(256), 0, stream>>>(Wv, w16 + 2 * E_ * E_, E_ * E_);
    copy_f32<<<dim3(4), dim3(256), 0, stream>>>(bq, biasP,          E_);
    copy_f32<<<dim3(4), dim3(256), 0, stream>>>(bk, biasP +     E_, E_);
    copy_f32<<<dim3(4), dim3(256), 0, stream>>>(bv, biasP + 2 * E_, E_);

    // 2) QKV projection: qkv(16384x3072) = x16 * w16^T + bias  (f16 out)
    wmma_gemm<true, false><<<dim3(F_ / 256, (B_ * S_) / 128, 1), dim3(256), 0, stream>>>(
        x16, 0, E_, w16, 0, E_, biasP, qkv16, 0, F_, E_, 1.0f);

    // 3) Transpose V -> vt (B, E, S)
    transpose_v<<<dim3(S_ / 32, E_ / 32, B_), dim3(32, 8), 0, stream>>>(qkv16, vt16);

    // 4) Scores: S[b] = Q[b] * K[b]^T / sqrt(E)   (M=N=2048, K=1024, batched, f16 out)
    wmma_gemm<false, false><<<dim3(S_ / 256, S_ / 128, B_), dim3(256), 0, stream>>>(
        qkv16 + 0, (size_t)S_ * F_, F_,
        qkv16 + E_, (size_t)S_ * F_, F_,
        nullptr, sc16, (size_t)S_ * S_, S_, E_, 0.03125f /* 1/sqrt(1024) */);

    // 5) masked_fill(1e-9) + softmax along keys, in place
    softmax_mask<<<dim3(B_ * S_), dim3(256), 0, stream>>>(sc16, mask);

    // 6) Output: out[b] = P[b] * V[b]  (= P * vt^T; M=2048, N=1024, K=2048, f32 out)
    wmma_gemm<false, true><<<dim3(E_ / 256, S_ / 128, B_), dim3(256), 0, stream>>>(
        sc16, (size_t)S_ * S_, S_,
        vt16, (size_t)E_ * S_, S_,
        nullptr, out, (size_t)S_ * E_, E_, S_, 1.0f);
}
